// SingleLayerDH_SRNN_86474871538208
// MI455X (gfx1250) — matrix operand order repaired and checked by reading, not compiled, not run on hardware
//
#include <hip/hip_runtime.h>
#include <hip/hip_bf16.h>

// Problem dims
constexpr int NB = 64;    // batch
constexpr int NS = 512;   // seq len
constexpr int NI = 256;   // input dim
constexpr int NH = 1024;  // hidden
constexpr int NO = 128;   // output

constexpr int KSLICE = 256;            // K elems staged in LDS per slice
constexpr int LDS_ROW = 264;           // 256 elems + 8 elem (16B) pad -> 528B row
constexpr int NSLICE = NH / KSLICE;    // 4

typedef __attribute__((ext_vector_type(16))) __bf16          v16bf;
typedef __attribute__((ext_vector_type(8)))  float           v8f;
typedef __attribute__((ext_vector_type(8)))  unsigned short  ushortx8;
typedef __attribute__((ext_vector_type(16))) unsigned short  ushortx16;
typedef __attribute__((ext_vector_type(4)))  unsigned int    uintx4;
typedef __attribute__((ext_vector_type(4)))  int             intx4;
typedef __attribute__((ext_vector_type(8)))  int             intx8;

// Tensor Data Mover availability (device pass only; host pass -> sync fallback)
#if defined(__AMDGCN__) && __has_builtin(__builtin_amdgcn_tensor_load_to_lds) && \
    __has_builtin(__builtin_amdgcn_s_wait_tensorcnt)
#define HAVE_TDM 1
#else
#define HAVE_TDM 0
#endif

// ---------------------------------------------------------------------------
// WMMA helpers (CDNA5 wave32, v_wmma_f32_16x16x32_bf16)
// ---------------------------------------------------------------------------
__device__ __forceinline__ v8f wmma_bf16(v16bf a, v16bf b, v8f c) {
  // (neg_a, A, neg_b, B, c_mod, C, reuse_a, reuse_b)
  return __builtin_amdgcn_wmma_f32_16x16x32_bf16(false, a, false, b, (short)0, c,
                                                 false, false);
}

// A-matrix 16x32 bf16 tile from a row-major [*, ld] matrix in GLOBAL memory.
// ISA layout: lanes 0-15 row M=lane hold K={0..7,16..23}; lanes 16-31 hold
// K={8..15,24..31} for the same rows. Two contiguous 16B chunks per lane.
__device__ __forceinline__ v16bf load_a_tile(const unsigned short* __restrict__ base,
                                             int ld, int m0, int k0, int lane) {
  const int m  = m0 + (lane & 15);
  const int kh = (lane >> 4) << 3;  // 0 or 8
  const unsigned short* p = base + (size_t)m * ld + k0 + kh;
  union { ushortx16 w16; struct { ushortx8 lo, hi; } h; } u;
  u.h.lo = *reinterpret_cast<const ushortx8*>(p);
  u.h.hi = *reinterpret_cast<const ushortx8*>(p + 16);
  return __builtin_bit_cast(v16bf, u.w16);
}

// Same A-tile but from the LDS-staged spike slice (padded rows, 528B stride).
// kk is the K offset within the slice (multiple of 32).
__device__ __forceinline__ v16bf lds_a_tile(const unsigned short* sm,
                                            int m0, int kk, int lane) {
  const int m  = m0 + (lane & 15);
  const int kh = (lane >> 4) << 3;
  const unsigned short* p = sm + m * LDS_ROW + kk + kh;
  union { ushortx16 w16; struct { ushortx8 lo, hi; } h; } u;
  u.h.lo = *reinterpret_cast<const ushortx8*>(p);        // ds_load_b128
  u.h.hi = *reinterpret_cast<const ushortx8*>(p + 16);
  return __builtin_bit_cast(v16bf, u.w16);
}

// B-matrix 32x16 bf16 tile, where B = W^T and W is row-major [N, ld].
// ISA layout: lanes 0-15 column N=lane hold K={0..7,16..23}; lanes 16-31 hold K={8..15,24..31}.
__device__ __forceinline__ v16bf load_b_tile(const unsigned short* __restrict__ wmat,
                                             int ld, int n0, int k0, int lane) {
  const int n  = n0 + (lane & 15);
  const int kb = k0 + ((lane >> 4) << 3);  // 0 or 8
  const unsigned short* p = wmat + (size_t)n * ld + kb;
  union { ushortx16 w16; struct { ushortx8 lo, hi; } h; } u;
  u.h.lo = *reinterpret_cast<const ushortx8*>(p);
  u.h.hi = *reinterpret_cast<const ushortx8*>(p + 16);
  return __builtin_bit_cast(v16bf, u.w16);
}

__device__ __forceinline__ unsigned short f2bf_rne(float f) {
  unsigned int u = __float_as_uint(f);
  u += 0x7FFFu + ((u >> 16) & 1u);
  return (unsigned short)(u >> 16);
}

// ---------------------------------------------------------------------------
// Stage one 64 x 256 bf16 K-slice of the spike matrix into LDS.
// TDM path: wave 0 issues tensor_load_to_lds with a 2D D# descriptor:
//   data_size=4B, tensor_dim0=512 dw (2048B memory row), tile=128x64 dw,
//   pad: every 128 dwords (one LDS row) insert 4 dwords -> 528B LDS row stride.
// Fallback: cooperative b128 copy into the identical padded layout.
// ---------------------------------------------------------------------------
#if HAVE_TDM
__device__ __forceinline__ void tdm_stage_slice(const unsigned short* gsp,
                                                unsigned lds_off, int ks) {
  const unsigned long long ga =
      (unsigned long long)gsp + (unsigned long long)ks * (KSLICE * 2);
  // D# group 0: count=1 | lds_addr | global_addr | type=2
  const uintx4 g0 = {1u, lds_off, (unsigned)ga,
                     (unsigned)(ga >> 32) | (2u << 30)};
  // D# group 1: data_size=4B, pad_enable, pad_interval=128dw, pad_amount=4dw
  const intx8 g1 = {
      (int)((2u << 16) | (1u << 20) | (6u << 22) | (3u << 25)),
      (int)(512u << 16),   // tensor_dim0 = 512 dwords (low16 in bits 63:48)
      (int)(64u << 16),    // tensor_dim0 hi=0 | tensor_dim1 = 64 rows
      (int)(128u << 16),   // tensor_dim1 hi=0 | tile_dim0 = 128 dwords
      64,                  // tile_dim1 = 64 | tile_dim2 = 0
      512,                 // tensor_dim0_stride = 512 dwords (2048B)
      0, 0};
  const intx4 gz = {0, 0, 0, 0};
#if __clang_major__ >= 23
  const intx8 gz8 = {0, 0, 0, 0, 0, 0, 0, 0};
  __builtin_amdgcn_tensor_load_to_lds(g0, g1, gz, gz, gz8, 0);
#else
  __builtin_amdgcn_tensor_load_to_lds(g0, g1, gz, gz, 0);
#endif
  __builtin_amdgcn_s_wait_tensorcnt(0);
}
#endif

__device__ __forceinline__ void stage_slice(const unsigned short* gsp,
                                            unsigned short* ssp, int ks,
                                            int tid, int wave) {
#if HAVE_TDM
  if (wave == 0) {
    tdm_stage_slice(gsp, (unsigned)(unsigned long long)(const void*)ssp, ks);
  }
#else
  (void)wave;
  // 64 rows x 32 chunks of 8 bf16; 2048 chunks over 256 threads
  for (int c = tid; c < 2048; c += 256) {
    const int m = c >> 5;
    const int ch = c & 31;
    const ushortx8 v = *reinterpret_cast<const ushortx8*>(
        gsp + (size_t)m * NH + ks * KSLICE + ch * 8);
    *reinterpret_cast<ushortx8*>(ssp + m * LDS_ROW + ch * 8) = v;
  }
#endif
}

// ---------------------------------------------------------------------------
// Prep kernels
// ---------------------------------------------------------------------------
__global__ void f32_to_bf16_kernel(const float* __restrict__ src,
                                   unsigned short* __restrict__ dst, int n) {
  int i = blockIdx.x * blockDim.x + threadIdx.x;
  if (i < n) dst[i] = f2bf_rne(src[i]);
}

__global__ void init_state_kernel(const float* __restrict__ tau_m,
                                  const float* __restrict__ tau_n,
                                  float* __restrict__ alpha_, float* __restrict__ beta_,
                                  float* __restrict__ d_st, float* __restrict__ mem_st,
                                  unsigned short* __restrict__ sbuf0) {
  int i = blockIdx.x * blockDim.x + threadIdx.x;
  if (i < NH) {
    alpha_[i] = 1.0f / (1.0f + __expf(-tau_m[i]));
    beta_[i]  = 1.0f / (1.0f + __expf(-tau_n[i]));
  }
  if (i < NB * NH) {
    d_st[i]   = 0.0f;
    mem_st[i] = 0.0f;
    sbuf0[i]  = 0;  // spike_{-1} = 0 (bf16 zero)
  }
}

// ---------------------------------------------------------------------------
// Phase 1: ff[s][b][h] = x[b][s][:] . W_in[h][:] + b_in[h]
// ---------------------------------------------------------------------------
__global__ __launch_bounds__(256) void ff_gemm_kernel(
    const unsigned short* __restrict__ xbf,    // [NB*NS, NI] bf16
    const unsigned short* __restrict__ winbf,  // [NH, NI] bf16
    const float* __restrict__ b_in,            // [NH]
    float* __restrict__ ff) {                  // [NS, NB, NH]
  const int lane = threadIdx.x & 31;
  const int wave = threadIdx.x >> 5;
  const int tM = blockIdx.x & ((NB * NS / 128) - 1);
  const int tN = blockIdx.x / (NB * NS / 128);
  const int m0 = tM * 128 + wave * 16;
  const int n0 = tN * 64;

  v8f acc[4] = {};
  for (int k0 = 0; k0 < NI; k0 += 32) {
    v16bf a = load_a_tile(xbf, NI, m0, k0, lane);
#pragma unroll
    for (int j = 0; j < 4; ++j) {
      v16bf b = load_b_tile(winbf, NI, n0 + j * 16, k0, lane);
      acc[j] = wmma_bf16(a, b, acc[j]);
    }
  }
  const int rbase = (lane >> 4) << 3;
  const int col = lane & 15;
#pragma unroll
  for (int j = 0; j < 4; ++j) {
    const int h = n0 + j * 16 + col;
    const float bi = b_in[h];
#pragma unroll
    for (int r = 0; r < 8; ++r) {
      const int row = m0 + rbase + r;  // = b*NS + s
      const int bb = row >> 9;
      const int ss = row & (NS - 1);
      ff[((size_t)ss * NB + bb) * NH + h] = acc[j][r] + bi;
    }
  }
}

// ---------------------------------------------------------------------------
// Phase 2: one launch per timestep (stream order = global sync).
// Blocks 0..15 : rec GEMM (64x1024x1024) + fused neuron update -> spike_t
// Block 16     : out_{t-1} = sigmoid(spike_{t-1} . W_out^T + b_out)
// Both stage spike_{t-1} K-slices into LDS via the TDM.
// ---------------------------------------------------------------------------
__global__ __launch_bounds__(256) void scan_step_kernel(
    int t,
    const float* __restrict__ ff,              // [NS, NB, NH]
    const unsigned short* __restrict__ wrec,   // [NH, NH] bf16
    const float* __restrict__ b_rec,
    const float* __restrict__ alpha_, const float* __restrict__ beta_,
    float* __restrict__ d_st, float* __restrict__ mem_st,   // [NB, NH]
    const unsigned short* __restrict__ spike_prev,          // [NB, NH] bf16
    unsigned short* __restrict__ spike_next,                // [NB, NH] bf16
    const unsigned short* __restrict__ wout,   // [NO, NH] bf16
    const float* __restrict__ b_out,
    float* __restrict__ out) {                 // [NB, NS, NO]
  __shared__ unsigned short ssp[NB * LDS_ROW];  // 33,792 B padded spike slice

  const int lane = threadIdx.x & 31;
  const int wave = threadIdx.x >> 5;
  const bool is_rec = (blockIdx.x < 16);
  if (is_rec && t >= NS) return;   // uniform per block
  if (!is_rec && t < 1) return;

  if (is_rec) {
    const int n0 = blockIdx.x * 64 + (wave >> 2) * 32;  // 2 n-tiles per wave
    const int m0 = (wave & 3) * 16;
    v8f acc0 = {}, acc1 = {};
    for (int ks = 0; ks < NSLICE; ++ks) {
      __syncthreads();  // previous slice fully consumed
      stage_slice(spike_prev, ssp, ks, threadIdx.x, wave);
      __syncthreads();  // LDS slice ready
#pragma unroll
      for (int kk = 0; kk < KSLICE; kk += 32) {
        const int k0 = ks * KSLICE + kk;
        v16bf a  = lds_a_tile(ssp, m0, kk, lane);
        v16bf b0 = load_b_tile(wrec, NH, n0,      k0, lane);
        v16bf b1 = load_b_tile(wrec, NH, n0 + 16, k0, lane);
        acc0 = wmma_bf16(a, b0, acc0);
        acc1 = wmma_bf16(a, b1, acc1);
      }
    }
    const int rbase = (lane >> 4) << 3;
    const int col = lane & 15;
    const float* __restrict__ fft = ff + (size_t)t * NB * NH;
#pragma unroll
    for (int j = 0; j < 2; ++j) {
      const v8f acc = j ? acc1 : acc0;
      const int h = n0 + j * 16 + col;
      const float al = alpha_[h];
      const float be = beta_[h];
      const float br = b_rec[h];
#pragma unroll
      for (int r = 0; r < 8; ++r) {
        const int bb = m0 + rbase + r;  // batch index
        const size_t sidx = (size_t)bb * NH + h;
        const float total = acc[r] + fft[sidx] + br;
        const float dd = be * d_st[sidx] + (1.0f - be) * total;
        float mm = al * mem_st[sidx] + (1.0f - al) * dd;
        const float sp = (mm > 1.0f) ? 1.0f : 0.0f;
        mm *= (1.0f - sp);
        d_st[sidx] = dd;
        mem_st[sidx] = mm;
        spike_next[sidx] = (sp > 0.0f) ? (unsigned short)0x3F80 : (unsigned short)0;
      }
    }
  } else {
    const int tp = t - 1;
    const int m0 = (wave & 3) * 16;
    const int n0 = (wave >> 2) * 64;
    v8f acc[4] = {};
    for (int ks = 0; ks < NSLICE; ++ks) {
      __syncthreads();
      stage_slice(spike_prev, ssp, ks, threadIdx.x, wave);
      __syncthreads();
#pragma unroll
      for (int kk = 0; kk < KSLICE; kk += 32) {
        const int k0 = ks * KSLICE + kk;
        v16bf a = lds_a_tile(ssp, m0, kk, lane);
#pragma unroll
        for (int j = 0; j < 4; ++j) {
          v16bf b = load_b_tile(wout, NH, n0 + j * 16, k0, lane);
          acc[j] = wmma_bf16(a, b, acc[j]);
        }
      }
    }
    const int rbase = (lane >> 4) << 3;
    const int col = lane & 15;
#pragma unroll
    for (int j = 0; j < 4; ++j) {
      const int o = n0 + j * 16 + col;
      const float bo = b_out[o];
#pragma unroll
      for (int r = 0; r < 8; ++r) {
        const int bb = m0 + rbase + r;
        const float v = acc[j][r] + bo;
        out[((size_t)bb * NS + tp) * NO + o] = 1.0f / (1.0f + __expf(-v));
      }
    }
  }
}

// ---------------------------------------------------------------------------
extern "C" void kernel_launch(void* const* d_in, const int* in_sizes, int n_in,
                              void* d_out, int out_size, void* d_ws, size_t ws_size,
                              hipStream_t stream) {
  (void)in_sizes; (void)n_in; (void)out_size; (void)ws_size;
  const float* x     = (const float*)d_in[0];  // [NB,NS,NI]
  const float* W_in  = (const float*)d_in[1];  // [NH,NI]
  const float* b_in  = (const float*)d_in[2];  // [NH]
  const float* W_rec = (const float*)d_in[3];  // [NH,NH]
  const float* b_rec = (const float*)d_in[4];  // [NH]
  const float* tau_m = (const float*)d_in[5];  // [NH]
  const float* tau_n = (const float*)d_in[6];  // [NH]
  const float* W_out = (const float*)d_in[7];  // [NO,NH]
  const float* b_out = (const float*)d_in[8];  // [NO]
  float* out = (float*)d_out;                  // [NB,NS,NO]

  char* ws = (char*)d_ws;
  size_t off = 0;
  auto carve = [&](size_t bytes) -> void* {
    void* p = ws + off;
    off += (bytes + 255) & ~(size_t)255;
    return p;
  };
  unsigned short* xbf    = (unsigned short*)carve((size_t)NB * NS * NI * 2);
  unsigned short* winbf  = (unsigned short*)carve((size_t)NH * NI * 2);
  unsigned short* wrecbf = (unsigned short*)carve((size_t)NH * NH * 2);
  unsigned short* woutbf = (unsigned short*)carve((size_t)NO * NH * 2);
  float*          alpha_ = (float*)carve((size_t)NH * 4);
  float*          beta_  = (float*)carve((size_t)NH * 4);
  float*          d_st   = (float*)carve((size_t)NB * NH * 4);
  float*          mem_st = (float*)carve((size_t)NB * NH * 4);
  unsigned short* sbuf0  = (unsigned short*)carve((size_t)NB * NH * 2);
  unsigned short* sbuf1  = (unsigned short*)carve((size_t)NB * NH * 2);
  float*          ff     = (float*)carve((size_t)NS * NB * NH * 4);

  {
    int n = NB * NS * NI;
    f32_to_bf16_kernel<<<(n + 255) / 256, 256, 0, stream>>>(x, xbf, n);
    n = NH * NI;
    f32_to_bf16_kernel<<<(n + 255) / 256, 256, 0, stream>>>(W_in, winbf, n);
    n = NH * NH;
    f32_to_bf16_kernel<<<(n + 255) / 256, 256, 0, stream>>>(W_rec, wrecbf, n);
    n = NO * NH;
    f32_to_bf16_kernel<<<(n + 255) / 256, 256, 0, stream>>>(W_out, woutbf, n);
    init_state_kernel<<<(NB * NH + 255) / 256, 256, 0, stream>>>(
        tau_m, tau_n, alpha_, beta_, d_st, mem_st, sbuf0);
  }

  // Phase 1: feedforward GEMM for all timesteps
  ff_gemm_kernel<<<(NB * NS / 128) * (NH / 64), 256, 0, stream>>>(xbf, winbf, b_in, ff);

  // Phase 2: sequential scan; spike ping-pong between sbuf0/sbuf1.
  unsigned short* sbuf[2] = {sbuf0, sbuf1};
  for (int t = 0; t <= NS; ++t) {
    scan_step_kernel<<<17, 256, 0, stream>>>(
        t, ff, wrecbf, b_rec, alpha_, beta_, d_st, mem_st,
        sbuf[t & 1], sbuf[(t + 1) & 1], woutbf, b_out, out);
  }
}
